// ConvMultiScaleAttention_56607668961288
// MI455X (gfx1250) — matrix operand order repaired and checked by reading, not compiled, not run on hardware
//
#include <hip/hip_runtime.h>

#define DIMC 256
#define NH   8
#define HD   32
#define LQ   2304          // 48*48
#define BB   4
#define OQKV 768
#define SCALE_F 0.17677669529663689f   // 1/sqrt(32)

typedef __attribute__((ext_vector_type(16))) __bf16 v16bf;
typedef __attribute__((ext_vector_type(8)))  float  v8f;
typedef __attribute__((ext_vector_type(4)))  unsigned u32x4;

union Frag { v16bf v; u32x4 u[2]; };

__device__ __forceinline__ unsigned short f2bf(float f) {
  union { float f; unsigned u; } x; x.f = f;
  unsigned r = x.u + 0x7FFFu + ((x.u >> 16) & 1u);   // round-to-nearest-even
  return (unsigned short)(r >> 16);
}

__device__ __forceinline__ u32x4 ld4(const unsigned short* p) {
  return *(const u32x4*)p;                           // 16B-aligned by construction
}

// ---------------- kernel 0: fp32 -> bf16 weight convert ----------------
__global__ void __launch_bounds__(256)
k_convert(const float* __restrict__ src, unsigned short* __restrict__ dst, int n) {
  int i = blockIdx.x * 256 + threadIdx.x;
  if (i < n) dst[i] = f2bf(src[i]);
}

// ---------------- kernel 0b: x (b,c,l) -> xT bf16 (b,l,c) ----------------
__global__ void __launch_bounds__(256)
k_xpose(const float* __restrict__ x, unsigned short* __restrict__ xT, int n) {
  int i = blockIdx.x * 256 + threadIdx.x;
  if (i >= n) return;
  int l = i % LQ;
  int c = (i / LQ) % DIMC;
  int b = i / (LQ * DIMC);
  xT[((size_t)b * LQ + l) * DIMC + c] = f2bf(x[i]);
}

// ---------------- kernel 1: QKV GEMM (768x256 @ 256xL) via bf16 WMMA ----------------
// Writes Q,K as [b*h][l][32]  (token-major) and V as [b*h][d][L] (dim-major).
__global__ void __launch_bounds__(256)
k_qkv(const unsigned short* __restrict__ wbf,
      const unsigned short* __restrict__ xT,
      const float* __restrict__ bqkv,
      unsigned short* __restrict__ Q,
      unsigned short* __restrict__ K,
      unsigned short* __restrict__ V) {
  const int wave = blockIdx.x * 8 + (threadIdx.x >> 5);
  const int lane = threadIdx.x & 31;
  const int lc = lane & 15, g = lane >> 4;
  const int lt = wave % (LQ / 16);
  const int ot = (wave / (LQ / 16)) % (OQKV / 16);
  const int b  = wave / ((LQ / 16) * (OQKV / 16));
  const int o0 = ot * 16, l0 = lt * 16;

  const unsigned short* wrow = wbf + (size_t)(o0 + lc) * DIMC;
  const unsigned short* xrow = xT + ((size_t)b * LQ + l0 + lc) * DIMC;

  v8f acc = {};
  #pragma unroll
  for (int c0 = 0; c0 < DIMC; c0 += 32) {
    Frag a, bf;
    a.u[0] = ld4(wrow + c0 + 8 * g);
    a.u[1] = ld4(wrow + c0 + 8 * g + 16);
    bf.u[0] = ld4(xrow + c0 + 8 * g);
    bf.u[1] = ld4(xrow + c0 + 8 * g + 16);
    acc = __builtin_amdgcn_wmma_f32_16x16x32_bf16(false, a.v, false, bf.v,
                                                  (short)0, acc, false, false);
  }

  const int r = o0 >> 8;                 // 0=q,1=k,2=v
  const int h = (o0 & 255) >> 5;
  const int dbase = o0 & 31;             // 0 or 16 (tile never crosses a head)
  const size_t bh = (size_t)b * NH + h;
  const int l = l0 + lc;
  #pragma unroll
  for (int v = 0; v < 8; ++v) {
    int row = v + 8 * g;
    int o = o0 + row;
    int d = dbase + row;
    unsigned short bv = f2bf(acc[v] + bqkv[o]);
    if (r == 0)      Q[(bh * LQ + l) * HD + d] = bv;
    else if (r == 1) K[(bh * LQ + l) * HD + d] = bv;
    else             V[(bh * HD + d) * LQ + l] = bv;
  }
}

// ---------------- kernel 2: flash attention, one 16-query tile per wave ----------------
__global__ void __launch_bounds__(256)
k_attn(const unsigned short* __restrict__ Q,
       const unsigned short* __restrict__ K,
       const unsigned short* __restrict__ V,
       unsigned short* __restrict__ O) {
  __shared__ __align__(16) unsigned short Pt[8][16][32];   // per-wave P transpose tile
  const int wv = threadIdx.x >> 5;
  const int wave = blockIdx.x * 8 + wv;
  const int lane = threadIdx.x & 31;
  const int lc = lane & 15, g = lane >> 4;
  const int qt = wave % (LQ / 16);
  const int h  = (wave / (LQ / 16)) % NH;
  const int b  = wave / ((LQ / 16) * NH);
  const size_t bh = (size_t)b * NH + h;

  const unsigned short* Qp = Q + (bh * LQ + qt * 16 + lc) * HD;
  const unsigned short* Kp = K + bh * LQ * HD;
  const unsigned short* Vp = V + bh * HD * LQ;

  Frag qa;                                   // A-frag of 16 queries x 32 dims
  qa.u[0] = ld4(Qp + 16 * g);
  qa.u[1] = ld4(Qp + 16 * g + 8);

  float m[8], ssum[8];
  v8f acc0 = {}, acc1 = {};
  #pragma unroll
  for (int v = 0; v < 8; ++v) { m[v] = -1e30f; ssum[v] = 0.f; }

  for (int j0 = 0; j0 < LQ; j0 += 32) {
    // ---- scores: S = Q x K^T (two 16-key B-frags) ----
    Frag kb0, kb1;
    const unsigned short* kr0 = Kp + (size_t)(j0 + lc) * HD + 16 * g;
    kb0.u[0] = ld4(kr0); kb0.u[1] = ld4(kr0 + 8);
    const unsigned short* kr1 = kr0 + 16 * HD;
    kb1.u[0] = ld4(kr1); kb1.u[1] = ld4(kr1 + 8);
    v8f z = {};
    v8f s0 = __builtin_amdgcn_wmma_f32_16x16x32_bf16(false, qa.v, false, kb0.v,
                                                     (short)0, z, false, false);
    v8f s1 = __builtin_amdgcn_wmma_f32_16x16x32_bf16(false, qa.v, false, kb1.v,
                                                     (short)0, z, false, false);
    // ---- online softmax (row stats live in 16-lane halves) ----
    float p0[8], p1[8];
    #pragma unroll
    for (int v = 0; v < 8; ++v) {
      float a0 = s0[v] * SCALE_F, a1 = s1[v] * SCALE_F;
      float rm = fmaxf(a0, a1);
      rm = fmaxf(rm, __shfl_xor(rm, 1));
      rm = fmaxf(rm, __shfl_xor(rm, 2));
      rm = fmaxf(rm, __shfl_xor(rm, 4));
      rm = fmaxf(rm, __shfl_xor(rm, 8));
      float mn = fmaxf(m[v], rm);
      float corr = __expf(m[v] - mn);
      p0[v] = __expf(a0 - mn);
      p1[v] = __expf(a1 - mn);
      float rs = p0[v] + p1[v];
      rs += __shfl_xor(rs, 1);
      rs += __shfl_xor(rs, 2);
      rs += __shfl_xor(rs, 4);
      rs += __shfl_xor(rs, 8);
      ssum[v] = ssum[v] * corr + rs;
      m[v] = mn;
      acc0[v] *= corr;
      acc1[v] *= corr;
    }
    // ---- transpose P (C/D layout -> A layout) through LDS ----
    #pragma unroll
    for (int v = 0; v < 8; ++v) {
      Pt[wv][v + 8 * g][lc]      = f2bf(p0[v]);
      Pt[wv][v + 8 * g][16 + lc] = f2bf(p1[v]);
    }
    Frag pa;
    pa.u[0] = *(const u32x4*)&Pt[wv][lc][16 * g];
    pa.u[1] = *(const u32x4*)&Pt[wv][lc][16 * g + 8];
    // ---- O += P x V^T (two 16-dim B-frags, contiguous keys) ----
    Frag vb0, vb1;
    const unsigned short* vr0 = Vp + (size_t)lc * LQ + j0 + 16 * g;
    vb0.u[0] = ld4(vr0); vb0.u[1] = ld4(vr0 + 8);
    const unsigned short* vr1 = vr0 + 16 * LQ;
    vb1.u[0] = ld4(vr1); vb1.u[1] = ld4(vr1 + 8);
    acc0 = __builtin_amdgcn_wmma_f32_16x16x32_bf16(false, pa.v, false, vb0.v,
                                                   (short)0, acc0, false, false);
    acc1 = __builtin_amdgcn_wmma_f32_16x16x32_bf16(false, pa.v, false, vb1.v,
                                                   (short)0, acc1, false, false);
  }
  // write O as [b][l][256] bf16 (token-major for the proj GEMM B-frags)
  #pragma unroll
  for (int v = 0; v < 8; ++v) {
    int l = qt * 16 + v + 8 * g;
    float inv = 1.0f / ssum[v];
    size_t base = ((size_t)b * LQ + l) * DIMC + h * HD;
    O[base + lc]      = f2bf(acc0[v] * inv);
    O[base + 16 + lc] = f2bf(acc1[v] * inv);
  }
}

// ---------------- kernel 3: output projection (256x256 @ 256xL) ----------------
__global__ void __launch_bounds__(256)
k_proj(const unsigned short* __restrict__ wp,
       const unsigned short* __restrict__ Ob,
       const float* __restrict__ bp,
       float* __restrict__ out) {
  const int wave = blockIdx.x * 8 + (threadIdx.x >> 5);
  const int lane = threadIdx.x & 31;
  const int lc = lane & 15, g = lane >> 4;
  const int lt = wave % (LQ / 16);
  const int ot = (wave / (LQ / 16)) % (DIMC / 16);
  const int b  = wave / ((LQ / 16) * (DIMC / 16));
  const int o0 = ot * 16, l0 = lt * 16;

  const unsigned short* wrow = wp + (size_t)(o0 + lc) * DIMC;
  const unsigned short* orow = Ob + ((size_t)b * LQ + l0 + lc) * DIMC;

  v8f acc = {};
  #pragma unroll
  for (int c0 = 0; c0 < DIMC; c0 += 32) {
    Frag a, bf;
    a.u[0] = ld4(wrow + c0 + 8 * g);
    a.u[1] = ld4(wrow + c0 + 8 * g + 16);
    bf.u[0] = ld4(orow + c0 + 8 * g);
    bf.u[1] = ld4(orow + c0 + 8 * g + 16);
    acc = __builtin_amdgcn_wmma_f32_16x16x32_bf16(false, a.v, false, bf.v,
                                                  (short)0, acc, false, false);
  }
  #pragma unroll
  for (int v = 0; v < 8; ++v) {
    int o = o0 + v + 8 * g;
    out[((size_t)b * DIMC + o) * LQ + l0 + lc] = acc[v] + bp[o];   // coalesced over lanes
  }
}

extern "C" void kernel_launch(void* const* d_in, const int* in_sizes, int n_in,
                              void* d_out, int out_size, void* d_ws, size_t ws_size,
                              hipStream_t stream) {
  const float* x      = (const float*)d_in[0];
  const float* w_qkv  = (const float*)d_in[1];
  const float* b_qkv  = (const float*)d_in[2];
  const float* w_proj = (const float*)d_in[3];
  const float* b_proj = (const float*)d_in[4];

  const size_t n_wqkv = (size_t)OQKV * DIMC;       // 196608
  const size_t n_wp   = (size_t)DIMC * DIMC;       // 65536
  const size_t n_x    = (size_t)BB * DIMC * LQ;    // 2359296
  const size_t n_head = (size_t)BB * NH * LQ * HD; // 2359296

  unsigned short* ws   = (unsigned short*)d_ws;    // ~24.1 MB total
  unsigned short* wqkv = ws;
  unsigned short* wpb  = wqkv + n_wqkv;
  unsigned short* xT   = wpb + n_wp;
  unsigned short* Qb   = xT + n_x;
  unsigned short* Kb   = Qb + n_head;
  unsigned short* Vb   = Kb + n_head;
  unsigned short* Ob   = Vb + n_head;

  k_convert<<<(int)((n_wqkv + 255) / 256), 256, 0, stream>>>(w_qkv, wqkv, (int)n_wqkv);
  k_convert<<<(int)((n_wp + 255) / 256), 256, 0, stream>>>(w_proj, wpb, (int)n_wp);
  k_xpose<<<(int)((n_x + 255) / 256), 256, 0, stream>>>(x, xT, (int)n_x);

  int waves1 = BB * (OQKV / 16) * (LQ / 16);   // 27648
  k_qkv<<<waves1 / 8, 256, 0, stream>>>(wqkv, xT, b_qkv, Qb, Kb, Vb);

  int waves2 = BB * NH * (LQ / 16);            // 4608
  k_attn<<<waves2 / 8, 256, 0, stream>>>(Qb, Kb, Vb, Ob);

  int waves3 = BB * (DIMC / 16) * (LQ / 16);   // 9216
  k_proj<<<waves3 / 8, 256, 0, stream>>>(wpb, Ob, b_proj, (float*)d_out);
}
